// LSTM_63711544869170
// MI455X (gfx1250) — matrix operand order repaired and checked by reading, not compiled, not run on hardware
//
#include <hip/hip_runtime.h>
#include <hip/hip_bf16.h>

// ---------------------------------------------------------------------------
// LSTM forward for MI455X (gfx1250), persistent-kernel design, round 3.
//   I=512, H=1024, B=64, S=512.
//   gates(64x4096) = [x_s | h_{t-1}] (64x1536, bf16) @ Wcat (1536x4096, bf16)
//   via v_wmma_f32_16x16x32_bf16 with f32 accumulation.
//   Operands staged in LDS via GLOBAL_LOAD_ASYNC_TO_LDS_B128 (ASYNCcnt),
//   double-buffered B k-slabs, 16 persistent WGs + monotonic global barrier.
// ---------------------------------------------------------------------------

typedef __attribute__((ext_vector_type(16))) __bf16 v16bf;
typedef __attribute__((ext_vector_type(8)))  __bf16 v8bf;
typedef __attribute__((ext_vector_type(8)))  float  v8f;
typedef __attribute__((ext_vector_type(4)))  int    v4i;

#define LSTM_I 512
#define LSTM_H 1024
#define LSTM_B 64
#define LSTM_S 512
#define LSTM_K (LSTM_I + LSTM_H)      // 1536
#define LSTM_N (4 * LSTM_H)           // 4096
#define KT_TOT (LSTM_K / 32)          // 48 k-tiles
#define NT_TOT (LSTM_N / 16)          // 256 n-tiles
#define NWG 16

// ---- async global->LDS path (gfx1250) -------------------------------------
#if defined(__has_builtin)
#if __has_builtin(__builtin_amdgcn_global_load_async_to_lds_b128)
#define USE_ASYNC_LDS 1
#endif
#endif
#ifndef USE_ASYNC_LDS
#define USE_ASYNC_LDS 0
#endif

typedef __attribute__((address_space(1))) v4i av4i_g;   // global int4
typedef __attribute__((address_space(3))) v4i av4i_l;   // LDS int4

__device__ __forceinline__ void cp16_g2l(__bf16* dst_lds, const __bf16* src_g) {
#if USE_ASYNC_LDS
    av4i_g* g = (av4i_g*)(unsigned long long)(uintptr_t)src_g;
    av4i_l* l = (av4i_l*)(unsigned)(uintptr_t)dst_lds;  // low 32 bits = LDS offset
    __builtin_amdgcn_global_load_async_to_lds_b128(g, l, 0, 0);
#else
    *(v8bf*)dst_lds = *(const v8bf*)src_g;
#endif
}

__device__ __forceinline__ void wait_async_lds() {
#if USE_ASYNC_LDS
#if __has_builtin(__builtin_amdgcn_s_wait_asynccnt)
    __builtin_amdgcn_s_wait_asynccnt(0);
#else
    asm volatile("s_wait_asynccnt 0x0" ::: "memory");
#endif
#endif
}

// ---- workspace layout (bytes) ---------------------------------------------
static constexpr size_t OFF_WBLK = 0;
static constexpr size_t SZ_WBLK  = (size_t)LSTM_K * LSTM_N * 2;            // 12 MB
static constexpr size_t OFF_XBF  = OFF_WBLK + SZ_WBLK;
static constexpr size_t SZ_XBF   = (size_t)LSTM_S * LSTM_B * LSTM_I * 2;   // 32 MB
static constexpr size_t OFF_BB   = OFF_XBF + SZ_XBF;
static constexpr size_t SZ_BB    = (size_t)LSTM_N * 4;
static constexpr size_t OFF_C    = OFF_BB + SZ_BB;
static constexpr size_t SZ_C     = (size_t)LSTM_B * LSTM_H * 4;
static constexpr size_t OFF_HB   = OFF_C + SZ_C;
static constexpr size_t SZ_HB    = (size_t)2 * LSTM_B * LSTM_H * 2;
static constexpr size_t OFF_BAR  = OFF_HB + SZ_HB;

// ---- dynamic LDS layout (bytes) -------------------------------------------
static constexpr unsigned SM_XH   = LSTM_B * LSTM_K * 2;        // 196608: [64][1536] bf16
static constexpr unsigned SM_B    = 2 * 16 * 512 * 2;           //  32768: double buffer, 16 tiles
static constexpr unsigned SM_TOT  = SM_XH + SM_B;               // 229376 <= 320KB/WGP

__device__ __forceinline__ float sigf(float x) {
    return 1.0f / (1.0f + __expf(-x));
}

// ---------------------------------------------------------------------------
__global__ void lstm_zero_init(float* __restrict__ c, __bf16* __restrict__ hb,
                               unsigned* __restrict__ bar) {
    int i = blockIdx.x * blockDim.x + threadIdx.x;
    if (i < LSTM_B * LSTM_H) c[i] = 0.0f;
    if (i < 2 * LSTM_B * LSTM_H) hb[i] = (__bf16)0.0f;
    if (i == 0) *bar = 0u;
}

// ---------------------------------------------------------------------------
// Pack Wcat (1536 x 4096 f32) -> bf16 WMMA B-operand tiles, grouped per WG:
//   wblk[kt][wg][t][lane][j], t = hl*4 + gate (hl = h-tile within WG),
//   lane = (n&15) + ((k&31) >= 16 ? 16 : 0),  j = (k&31) & 15.
// Each (kt, wg) slab is a contiguous 16 KB block -> clean async staging.
// ---------------------------------------------------------------------------
__global__ void lstm_pack_weights(__bf16* __restrict__ wblk,
                                  const float* wx0, const float* wx1,
                                  const float* wx2, const float* wx3,
                                  const float* wh0, const float* wh1,
                                  const float* wh2, const float* wh3) {
    size_t idx = (size_t)blockIdx.x * blockDim.x + threadIdx.x;
    if (idx >= (size_t)LSTM_K * LSTM_N) return;
    int k = (int)(idx >> 12);
    int n = (int)(idx & 4095);
    int gate = n >> 10, col = n & 1023;
    const float* wx[4] = {wx0, wx1, wx2, wx3};
    const float* wh[4] = {wh0, wh1, wh2, wh3};
    float w = (k < LSTM_I) ? wx[gate][(size_t)k * LSTM_H + col]
                           : wh[gate][(size_t)(k - LSTM_I) * LSTM_H + col];
    int kt = k >> 5, kk = k & 31;
    int nt = n >> 4;                 // 0..255
    int htile = nt & 63;             // h-tile 0..63
    int wg = htile >> 2;             // owning workgroup
    int t  = (htile & 3) * 4 + gate; // tile slot 0..15 inside WG slab
    int lane = (n & 15) + ((kk >= 16) ? 16 : 0);
    int j = kk & 15;
    size_t off = (((size_t)kt * NWG + wg) * 16 + t) * 512 + (size_t)lane * 16 + j;
    wblk[off] = (__bf16)w;
}

// Convert x (B,S,I) f32 -> xbf (S,B,I) bf16.
__global__ void lstm_pack_x(__bf16* __restrict__ xbf, const float* __restrict__ x) {
    size_t idx = (size_t)blockIdx.x * blockDim.x + threadIdx.x;
    if (idx >= (size_t)LSTM_B * LSTM_S * LSTM_I) return;
    int i = (int)(idx & 511);
    int s = (int)((idx >> 9) & 511);
    int b = (int)(idx >> 18);
    xbf[((size_t)s * LSTM_B + b) * LSTM_I + i] = (__bf16)x[idx];
}

__global__ void lstm_pack_bias(float* __restrict__ bb,
                               const float* bx0, const float* bx1,
                               const float* bx2, const float* bx3,
                               const float* bh0, const float* bh1,
                               const float* bh2, const float* bh3) {
    int n = blockIdx.x * blockDim.x + threadIdx.x;
    if (n >= LSTM_N) return;
    int gate = n >> 10, col = n & 1023;
    const float* bx[4] = {bx0, bx1, bx2, bx3};
    const float* bh[4] = {bh0, bh1, bh2, bh3};
    bb[n] = bx[gate][col] + bh[gate][col];
}

__device__ __forceinline__ v16bf cat8(v8bf lo, v8bf hi) {
    return __builtin_shufflevector(lo, hi, 0, 1, 2, 3, 4, 5, 6, 7,
                                   8, 9, 10, 11, 12, 13, 14, 15);
}

// ---------------------------------------------------------------------------
// Persistent recurrent kernel: 16 WGs x 512 threads (16 wave32 each).
// Per step: async-stage [x_s | h] (64x1536 bf16) into LDS + double-buffered
// 16 KB B k-slabs; each wave (mt, ht-local) does 4 gate tiles per k-tile with
// v_wmma_f32_16x16x32_bf16 reading both operands from LDS (ds_load_b128).
// ---------------------------------------------------------------------------
__global__ void __launch_bounds__(512, 1)
lstm_persistent(const __bf16* __restrict__ wblk,
                const __bf16* __restrict__ xbf,
                const float* __restrict__ bb,
                float* __restrict__ cst,
                __bf16* __restrict__ hb,
                unsigned* __restrict__ bar,
                float* __restrict__ out) {
    extern __shared__ __align__(16) char smem[];
    __bf16* XH = (__bf16*)smem;                 // [64][1536] row-major
    __bf16* Bs = (__bf16*)(smem + SM_XH);       // [2][16 tiles][32][16]

    const int tid  = threadIdx.x;
    const int lane = tid & 31;
    const int wid  = tid >> 5;                  // 0..15
    const int mt   = wid & 3;                   // M tile (rows mt*16..)
    const int htl  = wid >> 2;                  // h-tile within WG, 0..3
    const int ht   = (int)blockIdx.x * 4 + htl; // global h-tile 0..63
    const int l15  = lane & 15;
    const int kb   = (lane >> 4) << 4;          // A-operand K sub-offset (0/16)
    const int rowA = mt * 16 + l15;

    // staging assignment: 8 threads per batch row, 24 x 16B chunks each
    const int cr   = tid & 63;                  // batch row for XH staging
    const int slot = tid >> 6;                  // 0..7

    float bias[4];
#pragma unroll
    for (int g = 0; g < 4; ++g) bias[g] = bb[g * LSTM_H + ht * 16 + l15];

    for (int s = 0; s < LSTM_S; ++s) {
        const __bf16* hrd = hb + (size_t)(s & 1) * (LSTM_B * LSTM_H);
        __bf16* hwr = hb + (size_t)((s + 1) & 1) * (LSTM_B * LSTM_H);
        const __bf16* xs = xbf + (size_t)s * (LSTM_B * LSTM_I);

        // ---- stage [x_s | h] into LDS (async) ----
#pragma unroll 4
        for (int i = 0; i < 24; ++i) {
            int cu = (slot + 8 * i) * 8;        // u16 column 0..1528
            const __bf16* src = (cu < LSTM_I)
                ? (xs + (size_t)cr * LSTM_I + cu)
                : (hrd + (size_t)cr * LSTM_H + (cu - LSTM_I));
            cp16_g2l(&XH[(size_t)cr * LSTM_K + cu], src);
        }
        // ---- stage B slab kt=0 into buffer 0 ----
        {
            const __bf16* wsrc = wblk + (size_t)blockIdx.x * 8192;
            cp16_g2l(&Bs[tid * 16],     wsrc + tid * 16);
            cp16_g2l(&Bs[tid * 16 + 8], wsrc + tid * 16 + 8);
        }
        wait_async_lds();
        __syncthreads();

        v8f acc[4];
#pragma unroll
        for (int g = 0; g < 4; ++g) acc[g] = (v8f)(0.0f);

        for (int kt = 0; kt < KT_TOT; ++kt) {
            const int cur = kt & 1, nxt = cur ^ 1;

            // ---- async-prefetch next k-slab of B into the other buffer ----
            if (kt + 1 < KT_TOT) {
                const __bf16* wsrc =
                    wblk + ((size_t)(kt + 1) * NWG + blockIdx.x) * 8192;
                cp16_g2l(&Bs[nxt * 8192 + tid * 16],     wsrc + tid * 16);
                cp16_g2l(&Bs[nxt * 8192 + tid * 16 + 8], wsrc + tid * 16 + 8);
                if (kt + 2 < KT_TOT)
                    __builtin_prefetch(
                        wblk + ((size_t)(kt + 2) * NWG + blockIdx.x) * 8192
                             + tid * 16, 0, 1);
            }

            // ---- A tile (16x32 bf16) from LDS, contiguous k = 16*(lane/16) + s, as in B pack ----
            const __bf16* ap = &XH[(size_t)rowA * LSTM_K + kt * 32 + kb];
            v8bf a0 = *(const v8bf*)(ap);
            v8bf a1 = *(const v8bf*)(ap + 8);
            v16bf a = cat8(a0, a1);

#pragma unroll
            for (int g = 0; g < 4; ++g) {
                const __bf16* bp = &Bs[cur * 8192 + (htl * 4 + g) * 512 + lane * 16];
                v8bf b0 = *(const v8bf*)(bp);
                v8bf b1 = *(const v8bf*)(bp + 8);
                v16bf bmat = cat8(b0, b1);
                acc[g] = __builtin_amdgcn_wmma_f32_16x16x32_bf16(
                    false, a, false, bmat, (short)0, acc[g], false, false);
            }

            wait_async_lds();
            __syncthreads();    // buffer handoff + XH reuse safety
        }

        // ---- activations + state update on the 16x16 h block ----
        const int coln  = ht * 16 + l15;
        const int rbase = mt * 16 + ((lane >> 4) << 3);
#pragma unroll
        for (int v = 0; v < 8; ++v) {
            const int rown = rbase + v;
            const size_t cidx = (size_t)rown * LSTM_H + coln;
            float iv = sigf(acc[0][v] + bias[0]);
            float fv = sigf(acc[1][v] + bias[1]);
            float gv = tanhf(acc[2][v] + bias[2]);
            float ov = sigf(acc[3][v] + bias[3]);
            float cn = fv * cst[cidx] + iv * gv;
            cst[cidx] = cn;
            float hn = ov * tanhf(cn);
            hwr[cidx] = (__bf16)hn;
            if (s == LSTM_S - 1) {
                out[cidx] = hn;                              // h_T
                out[(size_t)LSTM_B * LSTM_H + cidx] = cn;    // c_T
            }
        }

        // ---- device-wide step barrier (monotonic counter) ----
        __threadfence();
        __syncthreads();
        if (tid == 0) {
            atomicAdd(bar, 1u);
            const unsigned target = (unsigned)NWG * (unsigned)(s + 1);
            while (*(volatile unsigned*)bar < target) {
                __builtin_amdgcn_s_sleep(1);
            }
        }
        __syncthreads();
        __threadfence();
    }
}

// ---------------------------------------------------------------------------
extern "C" void kernel_launch(void* const* d_in, const int* in_sizes, int n_in,
                              void* d_out, int out_size, void* d_ws, size_t ws_size,
                              hipStream_t stream) {
    (void)in_sizes; (void)n_in; (void)out_size; (void)ws_size;

    const float* x   = (const float*)d_in[0];
    const float* wii = (const float*)d_in[1];
    const float* bii = (const float*)d_in[2];
    const float* whi = (const float*)d_in[3];
    const float* bhi = (const float*)d_in[4];
    const float* wif = (const float*)d_in[5];
    const float* bif = (const float*)d_in[6];
    const float* whf = (const float*)d_in[7];
    const float* bhf = (const float*)d_in[8];
    const float* wig = (const float*)d_in[9];
    const float* big = (const float*)d_in[10];
    const float* whg = (const float*)d_in[11];
    const float* bhg = (const float*)d_in[12];
    const float* wio = (const float*)d_in[13];
    const float* bio = (const float*)d_in[14];
    const float* who = (const float*)d_in[15];
    const float* bho = (const float*)d_in[16];

    char* ws = (char*)d_ws;
    __bf16*   wblk = (__bf16*)(ws + OFF_WBLK);
    __bf16*   xbf  = (__bf16*)(ws + OFF_XBF);
    float*    bb   = (float*)(ws + OFF_BB);
    float*    cst  = (float*)(ws + OFF_C);
    __bf16*   hbuf = (__bf16*)(ws + OFF_HB);
    unsigned* bar  = (unsigned*)(ws + OFF_BAR);
    float*    out  = (float*)d_out;

    lstm_zero_init<<<dim3((2 * LSTM_B * LSTM_H + 255) / 256), dim3(256), 0, stream>>>(
        cst, hbuf, bar);

    lstm_pack_weights<<<dim3(((size_t)LSTM_K * LSTM_N + 255) / 256), dim3(256), 0, stream>>>(
        wblk, wii, wif, wig, wio, whi, whf, whg, who);

    lstm_pack_x<<<dim3(((size_t)LSTM_B * LSTM_S * LSTM_I + 255) / 256), dim3(256), 0, stream>>>(
        xbf, x);

    lstm_pack_bias<<<dim3((LSTM_N + 255) / 256), dim3(256), 0, stream>>>(
        bb, bii, bif, big, bio, bhi, bhf, bhg, bho);

    lstm_persistent<<<dim3(NWG), dim3(512), SM_TOT, stream>>>(
        wblk, xbf, bb, cst, hbuf, bar, out);
}
